// SIMCLR_43301860278640
// MI455X (gfx1250) — compile-verified, matrix operand order checked by reading, not run on hardware
//
#include <hip/hip_runtime.h>
#include <hip/hip_bf16.h>

// ---- problem constants (match reference) ----
#define C_CH   20      // node channels
#define D_EA   25      // edge-attr dim
#define ZIN    65      // 2C + D
#define KP     96      // K padded to 3 x 32 for bf16 WMMA
#define NOUT   40      // fused [Wf | Ws] output cols
#define NP     48      // N padded to 3 x 16
#define TILE_E 128     // edges per block-tile (8 waves x 16 edges)
#define ZSTR   104     // LDS row stride (bf16): 208 B rows (16B multiple)
#define EPSBN  1e-5f

typedef __attribute__((ext_vector_type(16))) __bf16 v16bf;
typedef __attribute__((ext_vector_type(8)))  __bf16 v8bf;
typedef __attribute__((ext_vector_type(8)))  float  v8f;
typedef __attribute__((ext_vector_type(4)))  float  v4f;

// hardware approximate reciprocal: single v_rcp_f32 (TRANS pipe, co-issues
// with the v_exp/v_log already in the epilogue) instead of IEEE div sequence
__device__ __forceinline__ float fast_rcp(float x) {
    return __builtin_amdgcn_rcpf(x);
}

// ---------------- generic helpers ----------------
__global__ void simclr_zero_kernel(float* __restrict__ p, int n) {
    for (int i = blockIdx.x * blockDim.x + threadIdx.x; i < n;
         i += gridDim.x * blockDim.x)
        p[i] = 0.f;
}

// agg = h (residual init), hbf = bf16(h)
__global__ void simclr_prep_kernel(const float* __restrict__ h,
                                   float* __restrict__ agg,
                                   __bf16* __restrict__ hbf, int n) {
    int i = blockIdx.x * blockDim.x + threadIdx.x;
    if (i >= n) return;
    float v = h[i];
    agg[i] = v;
    hbf[i] = (__bf16)v;
}

// ---------------- edge message kernel (WMMA, persistent) ----------------
// Grid-stride over 128-edge tiles. Per tile:
//   z[128 x 96](bf16,LDS) @ Wcat[96 x 48](bf16,LDS, pre-swizzled to the
//   per-lane WMMA fragment layout) via v_wmma_f32_16x16x32_bf16
//   (3 k-chunks x 3 n-tiles per wave), f32 staging in LDS as contiguous
//   per-lane v8f, then gate*softplus epilogue + atomic scatter to agg[dst].
// All fragment traffic is ds_load_b128 / ds_store_b128 / ds_store_b64.
__global__ void __launch_bounds__(256)
simclr_edge_msg_kernel(const __bf16* __restrict__ hbf,
                       const float* __restrict__ eattr,
                       const int* __restrict__ src,
                       const int* __restrict__ dst,
                       const float* __restrict__ Wf,
                       const float* __restrict__ Ws,
                       const float* __restrict__ bfv,
                       const float* __restrict__ bsv,
                       float* __restrict__ agg,
                       int Etot) {
    __shared__ __attribute__((aligned(16))) __bf16 wswz[9 * 32 * 16]; //  9216 B
    __shared__ float blds[NP];                                        //   192 B
    __shared__ __attribute__((aligned(16))) __bf16 zlds[TILE_E * ZSTR]; // 26624 B
    __shared__ __attribute__((aligned(16))) float  mlds[8 * 3 * 32 * 8]; // 24576 B

    const int tid   = threadIdx.x;
    const int lane  = tid & 31;
    const int w     = tid >> 5;              // wave 0..7
    const int khalf = (lane < 16) ? 0 : 8;   // ISA 16-bit A/B half-K base
    const int nIdx  = lane & 15;

    // ---- weights -> LDS, pre-swizzled per (nt, kc, lane, half) so each
    //      lane's 16 B-halves are contiguous (two ds_load_b128 later) ----
    for (int i = tid; i < 9 * 32 * 16; i += 256) {
        int h   = i & 15;
        int ln  = (i >> 4) & 31;
        int seg = i >> 9;                  // nt*3 + kc, 0..8
        int kc  = seg % 3;
        int nt  = seg / 3;
        int kh  = (ln < 16) ? 0 : 8;
        int k   = kc * 32 + kh + ((h >> 3) << 4) + (h & 7);
        int j   = nt * 16 + (ln & 15);
        float wv = 0.f;
        if (k < ZIN && j < NOUT)
            wv = (j < C_CH) ? Wf[k * C_CH + j] : Ws[k * C_CH + (j - C_CH)];
        wswz[i] = (__bf16)wv;
    }
    if (tid < NP)
        blds[tid] = (tid < C_CH) ? bfv[tid]
                  : ((tid < NOUT) ? bsv[tid - C_CH] : 0.f);

    const int ntiles = (Etot + TILE_E - 1) / TILE_E;
    for (int tile = blockIdx.x; tile < ntiles; tile += gridDim.x) {
        const int e0 = tile * TILE_E;

        // gfx1250 prefetch (global_prefetch_b8): this block's next tile of
        // the streamed edge_attr (the only true-HBM stream in this kernel)
        {
            long long pe = (long long)(tile + gridDim.x) * TILE_E;
            if (pe < Etot)
                __builtin_prefetch(eattr + (size_t)pe * D_EA + tid * 50, 0, 3);
        }

        // ---- gather z = [x[dst] || x[src] || e], 4 bf16 per ds_store_b64 ----
        for (int i = tid; i < TILE_E * (KP / 4); i += 256) {
            int el = i / (KP / 4);
            int q  = i % (KP / 4);           // features 4q..4q+3
            int eg = e0 + el;
            __bf16 vals[4] = {(__bf16)0.f, (__bf16)0.f, (__bf16)0.f, (__bf16)0.f};
            if (eg < Etot) {
                int f0 = q * 4;
                if (f0 < C_CH) {                         // x[dst] (8B aligned)
                    int d = dst[eg];
                    __builtin_memcpy(vals, hbf + (size_t)d * C_CH + f0, 8);
                } else if (f0 < 2 * C_CH) {              // x[src]
                    int s = src[eg];
                    __builtin_memcpy(vals, hbf + (size_t)s * C_CH + (f0 - C_CH), 8);
                } else if (f0 < ZIN + 3) {               // edge_attr (f32->bf16)
#pragma unroll
                    for (int t = 0; t < 4; ++t) {
                        int f = f0 + t;
                        if (f < ZIN)
                            vals[t] = (__bf16)eattr[(size_t)eg * D_EA + (f - 2 * C_CH)];
                    }
                }
            }
            __builtin_memcpy(&zlds[el * ZSTR + q * 4], vals, 8);
        }
        __syncthreads();

        // ---- per-wave WMMA: 16 edges x 48 outputs ----
        // A fragment: two contiguous 8-half runs -> 2 x ds_load_b128
        v16bf afrag[3];
#pragma unroll
        for (int kc = 0; kc < 3; ++kc) {
            const v8bf* zp =
                (const v8bf*)&zlds[(w * 16 + nIdx) * ZSTR + kc * 32 + khalf];
            v8bf lo = zp[0];   // K halves 0..7
            v8bf hi = zp[2];   // K halves 16..23
            afrag[kc] = __builtin_shufflevector(
                lo, hi, 0, 1, 2, 3, 4, 5, 6, 7, 8, 9, 10, 11, 12, 13, 14, 15);
        }

#pragma unroll
        for (int nt = 0; nt < 3; ++nt) {
            v8f acc = {0.f, 0.f, 0.f, 0.f, 0.f, 0.f, 0.f, 0.f};
#pragma unroll
            for (int kc = 0; kc < 3; ++kc) {
                const v8bf* wp =
                    (const v8bf*)&wswz[(((nt * 3 + kc) * 32) + lane) * 16];
                v8bf lo = wp[0];
                v8bf hi = wp[1];
                v16bf b = __builtin_shufflevector(
                    lo, hi, 0, 1, 2, 3, 4, 5, 6, 7, 8, 9, 10, 11, 12, 13, 14, 15);
                acc = __builtin_amdgcn_wmma_f32_16x16x32_bf16(
                    false, afrag[kc], false, b, (short)0, acc, false, false);
            }
            // stage D tile as contiguous per-lane v8f (2 x ds_store_b128)
            float* mp = &mlds[(((w * 3 + nt) * 32) + lane) * 8];
            *(v4f*)mp       = (v4f){acc[0], acc[1], acc[2], acc[3]};
            *(v4f*)(mp + 4) = (v4f){acc[4], acc[5], acc[6], acc[7]};
        }
        __syncthreads();

        // ---- epilogue: msg = sigmoid(zWf+bf) * softplus(zWs+bs) ----
        // D layout: value (tile nt, row M, col N) lives at
        // mlds[((w*3+nt)*32 + N + (M>=8?16:0))*8 + (M&7)]
        for (int i = tid; i < TILE_E * C_CH; i += 256) {
            int el = i / C_CH, c = i % C_CH;
            int eg = e0 + el;
            if (eg < Etot) {
                int w_   = el >> 4;
                int mrow = el & 15;
                int ln   = (mrow & 8) ? 16 : 0;
                int r    = mrow & 7;
                int ntf  = c >> 4;
                int fidx = (((w_ * 3 + ntf) * 32) + (ln + (c & 15))) * 8 + r;
                int cs   = c + C_CH;
                int nts  = cs >> 4;
                int sidx = (((w_ * 3 + nts) * 32) + (ln + (cs & 15))) * 8 + r;
                float f = mlds[fidx] + blds[c];
                float s = mlds[sidx] + blds[C_CH + c];
                float gate = fast_rcp(1.f + __expf(-f));       // v_rcp_f32
                float sp = (s > 0.f) ? (s + __logf(1.f + __expf(-s)))
                                     : __logf(1.f + __expf(s));
                atomicAdd(&agg[(size_t)dst[eg] * C_CH + c], gate * sp);
            }
        }
        // loop-carried LDS hazards are covered by the two __syncthreads above
    }
}

// ---------------- batchnorm ----------------
// stats[0..19]=sum, [20..39]=sumsq, [40..59]=scale, [60..79]=shift
__global__ void simclr_bn_stats_kernel(const float* __restrict__ h,
                                       float* __restrict__ stats, int Nrows) {
    __shared__ float ssum[32], ssq[32];
    int tid = threadIdx.x;
    if (tid < 32) { ssum[tid] = 0.f; ssq[tid] = 0.f; }
    __syncthreads();
    int c  = tid & 31;
    int rg = tid >> 5;  // 8 row-groups per block
    if (c < C_CH) {
        float s = 0.f, q = 0.f;
        for (int r = blockIdx.x * 8 + rg; r < Nrows; r += gridDim.x * 8) {
            float v = h[(size_t)r * C_CH + c];
            s += v; q += v * v;
        }
        atomicAdd(&ssum[c], s);   // ds_add_f32
        atomicAdd(&ssq[c], q);
    }
    __syncthreads();
    if (tid < C_CH) {
        atomicAdd(&stats[tid], ssum[tid]);
        atomicAdd(&stats[C_CH + tid], ssq[tid]);
    }
}

__global__ void simclr_bn_finalize_kernel(float* __restrict__ stats,
                                          const float* __restrict__ gamma,
                                          const float* __restrict__ beta,
                                          float invN) {
    int c = threadIdx.x;
    if (c < C_CH) {
        float mu    = stats[c] * invN;
        float var   = stats[C_CH + c] * invN - mu * mu;
        float scale = rsqrtf(var + EPSBN) * gamma[c];
        stats[40 + c] = scale;
        stats[60 + c] = beta[c] - mu * scale;
    }
}

// hout = agg*scale + shift; also refresh bf16 copy for next layer's gathers
__global__ void simclr_bn_apply_kernel(const float* __restrict__ agg,
                                       const float* __restrict__ stats,
                                       float* __restrict__ hout,
                                       __bf16* __restrict__ hbf, int n) {
    int i = blockIdx.x * blockDim.x + threadIdx.x;
    if (i >= n) return;
    int c = i % C_CH;
    float v = agg[i] * stats[40 + c] + stats[60 + c];
    hout[i] = v;
    hbf[i]  = (__bf16)v;
}

// ---------------- pooling + head ----------------
__global__ void simclr_pool_kernel(const float* __restrict__ h,
                                   const int* __restrict__ batch,
                                   float* __restrict__ pooled,
                                   float* __restrict__ counts, int n) {
    int i = blockIdx.x * blockDim.x + threadIdx.x;
    if (i >= n) return;
    int node = i / C_CH, c = i % C_CH;
    int g = batch[node];
    atomicAdd(&pooled[(size_t)g * C_CH + c], h[i]);
    if (c == 0) atomicAdd(&counts[g], 1.f);
}

__global__ void simclr_head1_kernel(const float* __restrict__ pooled,
                                    const float* __restrict__ counts,
                                    const float* __restrict__ W1,
                                    const float* __restrict__ c1,
                                    float* __restrict__ t, int n) {
    int i = blockIdx.x * blockDim.x + threadIdx.x;
    if (i >= n) return;
    int g = i / C_CH, j = i % C_CH;
    float cnt = counts[g];
    float inv = fast_rcp(cnt > 1.f ? cnt : 1.f);
    float acc = c1[j];
#pragma unroll
    for (int k = 0; k < C_CH; ++k)
        acc += pooled[g * C_CH + k] * inv * W1[k * C_CH + j];
    t[i] = (acc > 0.f) ? acc : 0.01f * acc;  // leaky_relu(0.01)
}

__global__ void simclr_head2_kernel(const float* __restrict__ t,
                                    const float* __restrict__ W2,
                                    const float* __restrict__ c2,
                                    float* __restrict__ y, int n) {
    int i = blockIdx.x * blockDim.x + threadIdx.x;
    if (i >= n) return;
    int g = i / C_CH, j = i % C_CH;
    float acc = c2[j];
#pragma unroll
    for (int k = 0; k < C_CH; ++k)
        acc += t[g * C_CH + k] * W2[k * C_CH + j];
    y[i] = acc;
}

// ---------------- launcher ----------------
extern "C" void kernel_launch(void* const* d_in, const int* in_sizes, int n_in,
                              void* d_out, int out_size, void* d_ws,
                              size_t ws_size, hipStream_t stream) {
    const float* x     = (const float*)d_in[0];
    const int*   ei    = (const int*)d_in[1];
    const float* eattr = (const float*)d_in[2];
    const int*   batch = (const int*)d_in[3];
    const float* Wf0 = (const float*)d_in[4];
    const float* bf0 = (const float*)d_in[5];
    const float* Ws0 = (const float*)d_in[6];
    const float* bs0 = (const float*)d_in[7];
    const float* g0  = (const float*)d_in[8];
    const float* be0 = (const float*)d_in[9];
    const float* Wf1 = (const float*)d_in[10];
    const float* bf1 = (const float*)d_in[11];
    const float* Ws1 = (const float*)d_in[12];
    const float* bs1 = (const float*)d_in[13];
    const float* g1  = (const float*)d_in[14];
    const float* be1 = (const float*)d_in[15];
    const float* W1  = (const float*)d_in[16];
    const float* c1  = (const float*)d_in[17];
    const float* W2  = (const float*)d_in[18];
    const float* c2  = (const float*)d_in[19];

    const int N = in_sizes[0] / C_CH;
    const int E = in_sizes[1] / 2;
    const int G = out_size / C_CH;
    const int* srcI = ei;       // edge_index[0]
    const int* dstI = ei + E;   // edge_index[1]

    // workspace layout
    char* ws = (char*)d_ws;
    float*  agg    = (float*)ws;  ws += (size_t)N * C_CH * sizeof(float);
    float*  hA     = (float*)ws;  ws += (size_t)N * C_CH * sizeof(float);
    __bf16* hbf    = (__bf16*)ws; ws += (size_t)N * C_CH * sizeof(__bf16);
    float*  stats  = (float*)ws;  ws += 80 * sizeof(float);
    float*  pooled = (float*)ws;  ws += (size_t)G * C_CH * sizeof(float);
    float*  counts = (float*)ws;  ws += (size_t)G * sizeof(float);   // contiguous after pooled
    float*  thid   = (float*)ws;

    const int   nc      = N * C_CH;
    const int   gnc     = (nc + 255) / 256;
    const int   eblocks = (E + TILE_E - 1) / TILE_E;
    const int   egrid   = eblocks < 4096 ? eblocks : 4096;  // persistent tiles
    const float invN    = 1.0f / (float)N;

    // ---- layer 0 ----
    simclr_prep_kernel<<<gnc, 256, 0, stream>>>(x, agg, hbf, nc);
    simclr_zero_kernel<<<1, 128, 0, stream>>>(stats, 80);
    simclr_edge_msg_kernel<<<egrid, 256, 0, stream>>>(
        hbf, eattr, srcI, dstI, Wf0, Ws0, bf0, bs0, agg, E);
    simclr_bn_stats_kernel<<<512, 256, 0, stream>>>(agg, stats, N);
    simclr_bn_finalize_kernel<<<1, 32, 0, stream>>>(stats, g0, be0, invN);
    simclr_bn_apply_kernel<<<gnc, 256, 0, stream>>>(agg, stats, hA, hbf, nc);

    // ---- layer 1 ----
    simclr_prep_kernel<<<gnc, 256, 0, stream>>>(hA, agg, hbf, nc);
    simclr_zero_kernel<<<1, 128, 0, stream>>>(stats, 80);
    simclr_edge_msg_kernel<<<egrid, 256, 0, stream>>>(
        hbf, eattr, srcI, dstI, Wf1, Ws1, bf1, bs1, agg, E);
    simclr_bn_stats_kernel<<<512, 256, 0, stream>>>(agg, stats, N);
    simclr_bn_finalize_kernel<<<1, 32, 0, stream>>>(stats, g1, be1, invN);
    simclr_bn_apply_kernel<<<gnc, 256, 0, stream>>>(agg, stats, hA, hbf, nc);

    // ---- mean pool + projection head ----
    simclr_zero_kernel<<<(G * C_CH + G + 127) / 128, 128, 0, stream>>>(
        pooled, G * C_CH + G);  // pooled and counts are contiguous
    simclr_pool_kernel<<<gnc, 256, 0, stream>>>(hA, batch, pooled, counts, nc);
    simclr_head1_kernel<<<(G * C_CH + 255) / 256, 256, 0, stream>>>(
        pooled, counts, W1, c1, thid, G * C_CH);
    simclr_head2_kernel<<<(G * C_CH + 255) / 256, 256, 0, stream>>>(
        thid, W2, c2, (float*)d_out, G * C_CH);
}
